// GNN_55250459296217
// MI455X (gfx1250) — matrix operand order repaired and checked by reading, not compiled, not run on hardware
//
#include <hip/hip_runtime.h>

#define N_NODES 100000
#define NPAD    100096            // multiple of 128 (block row tile)
#define N_EDGES 200000
#define EMB     300
#define EMB2    600
#define NUM_LAYER 5
#define KP1     320               // padded K for GEMM1 (10 chunks of 32)
#define KT1     10
#define CT1     40                // col tiles for GEMM1 (640 padded cols)
#define COLP1   640               // hid padded cols == padded K for GEMM2
#define KT2     20
#define CT2     20                // col tiles for GEMM2 (320 padded cols)
#define COLP2   320
#define BN_EPS  1e-5f

typedef __attribute__((ext_vector_type(16))) __bf16 v16bf;
typedef __attribute__((ext_vector_type(8)))  float  v8f;

union AFrag { v16bf v; uint4 q[2]; unsigned int u[8]; };

__device__ __forceinline__ unsigned short f2bf(float f) {
    unsigned int u = __builtin_bit_cast(unsigned int, f);
    u += 0x7FFFu + ((u >> 16) & 1u);          // round-to-nearest-even
    return (unsigned short)(u >> 16);
}

// ---------------- h0 = atom_emb1[x0] + atom_emb2[x1] ----------------
__global__ void k_embed(const int* __restrict__ x,
                        const float* __restrict__ e1,
                        const float* __restrict__ e2,
                        float* __restrict__ h) {
    long long i = (long long)blockIdx.x * blockDim.x + threadIdx.x;
    if (i >= (long long)N_NODES * EMB) return;
    int n = (int)(i / EMB), d = (int)(i % EMB);
    h[i] = e1[x[2 * n] * EMB + d] + e2[x[2 * n + 1] * EMB + d];
}

// ---------------- message + scatter-add (one edge per block) ----------------
__global__ void k_scatter(const int* __restrict__ ei, const int* __restrict__ ea,
                          const float* __restrict__ ee1, const float* __restrict__ ee2,
                          const float* __restrict__ h, float* __restrict__ agg) {
    int e = blockIdx.x;
    int d = threadIdx.x;
    if (d >= EMB) return;
    int row, col, a0, a1;
    if (e < N_EDGES) {
        row = ei[e]; col = ei[N_EDGES + e];
        a0 = ea[2 * e]; a1 = ea[2 * e + 1];
    } else {                       // self loop, edge_attr = (4, 0)
        row = col = e - N_EDGES; a0 = 4; a1 = 0;
    }
    float m = h[(long long)row * EMB + d] + ee1[a0 * EMB + d] + ee2[a1 * EMB + d];
    unsafeAtomicAdd(&agg[(size_t)col * KP1 + d], m);
}

// ---------------- agg fp32 [NPAD][KP1] -> bf16 (packed pairs) ----------------
__global__ void k_convA(const float* __restrict__ agg, unsigned int* __restrict__ aggb) {
    long long i = (long long)blockIdx.x * blockDim.x + threadIdx.x;   // one bf16 pair
    if (i >= (long long)NPAD * KP1 / 2) return;
    float2 f = ((const float2*)agg)[i];
    aggb[i] = (unsigned int)f2bf(f.x) | ((unsigned int)f2bf(f.y) << 16);
}

// ---------------- pack W1 into B-fragment-ready layout ----------------
// Bpk dword index: ((kt*CT1 + ct)*32 + lane)*8 + j ; pair K = kt*32 + 8*g + 2*j + (j >= 4 ? 8 : 0)
__global__ void k_packW1(const float* __restrict__ W1, unsigned int* __restrict__ Bpk) {
    int i = blockIdx.x * blockDim.x + threadIdx.x;
    if (i >= KT1 * CT1 * 256) return;
    int j = i & 7, lane = (i >> 3) & 31;
    int ct = (i >> 8) % CT1, kt = (i >> 8) / CT1;
    int g = lane >> 4, n = ct * 16 + (lane & 15);
    int k = kt * 32 + 8 * g + 2 * j + (j >= 4 ? 8 : 0);
    unsigned short lo = (n < EMB2 && k     < EMB) ? f2bf(W1[n * EMB + k])     : (unsigned short)0;
    unsigned short hi = (n < EMB2 && k + 1 < EMB) ? f2bf(W1[n * EMB + k + 1]) : (unsigned short)0;
    Bpk[i] = (unsigned int)lo | ((unsigned int)hi << 16);
}

__global__ void k_packW2(const float* __restrict__ W2, unsigned int* __restrict__ Bpk) {
    int i = blockIdx.x * blockDim.x + threadIdx.x;
    if (i >= KT2 * CT2 * 256) return;
    int j = i & 7, lane = (i >> 3) & 31;
    int ct = (i >> 8) % CT2, kt = (i >> 8) / CT2;
    int g = lane >> 4, n = ct * 16 + (lane & 15);
    int k = kt * 32 + 8 * g + 2 * j + (j >= 4 ? 8 : 0);
    unsigned short lo = (n < EMB && k     < EMB2) ? f2bf(W2[n * EMB2 + k])     : (unsigned short)0;
    unsigned short hi = (n < EMB && k + 1 < EMB2) ? f2bf(W2[n * EMB2 + k + 1]) : (unsigned short)0;
    Bpk[i] = (unsigned int)lo | ((unsigned int)hi << 16);
}

// ---------------- GEMM1: hid = relu(agg @ W1^T + b1), bf16 out ----------------
// block = 8 waves, each wave owns a 16(M) x 64(N) strip; block tile 128 x 64.
__global__ void __launch_bounds__(256)
k_gemm1(const unsigned short* __restrict__ A, const unsigned int* __restrict__ Bpk,
        const float* __restrict__ bias, unsigned short* __restrict__ hid) {
    const int lane = threadIdx.x & 31;
    const int wq   = threadIdx.x >> 5;
    const int g    = lane >> 4, ln = lane & 15;
    const int rowBase = blockIdx.y * 128 + wq * 16;
    const int ctBase  = blockIdx.x * 4;
    const unsigned short* arow = A + (size_t)(rowBase + ln) * KP1;
    v8f acc[4] = {};
    for (int kt = 0; kt < KT1; ++kt) {
        AFrag a;
        a.q[0] = *(const uint4*)(arow + kt * 32 + 8 * g);
        a.q[1] = *(const uint4*)(arow + kt * 32 + 16 + 8 * g);
        const unsigned int* bb = Bpk + ((size_t)(kt * CT1 + ctBase) * 32 + lane) * 8;
        #pragma unroll
        for (int ct = 0; ct < 4; ++ct) {
            AFrag b;
            b.q[0] = *(const uint4*)(bb + ct * 256);
            b.q[1] = *(const uint4*)(bb + ct * 256 + 4);
            acc[ct] = __builtin_amdgcn_wmma_f32_16x16x32_bf16(false, a.v, false, b.v,
                                                              (short)0, acc[ct], false, false);
        }
    }
    #pragma unroll
    for (int ct = 0; ct < 4; ++ct) {
        int col = (ctBase + ct) * 16 + ln;
        float bv = (col < EMB2) ? bias[col] : 0.f;
        #pragma unroll
        for (int j = 0; j < 8; ++j) {
            int row = rowBase + 8 * g + j;
            float v = acc[ct][j] + bv;
            v = v > 0.f ? v : 0.f;
            hid[(size_t)row * COLP1 + col] = (col < EMB2) ? f2bf(v) : (unsigned short)0;
        }
    }
}

// ---------------- GEMM2: h = hid @ W2^T + b2, fp32 out ----------------
__global__ void __launch_bounds__(256)
k_gemm2(const unsigned short* __restrict__ A, const unsigned int* __restrict__ Bpk,
        const float* __restrict__ bias, float* __restrict__ out) {
    const int lane = threadIdx.x & 31;
    const int wq   = threadIdx.x >> 5;
    const int g    = lane >> 4, ln = lane & 15;
    const int rowBase = blockIdx.y * 128 + wq * 16;
    const int ctBase  = blockIdx.x * 4;
    const unsigned short* arow = A + (size_t)(rowBase + ln) * COLP1;
    v8f acc[4] = {};
    for (int kt = 0; kt < KT2; ++kt) {
        AFrag a;
        a.q[0] = *(const uint4*)(arow + kt * 32 + 8 * g);
        a.q[1] = *(const uint4*)(arow + kt * 32 + 16 + 8 * g);
        const unsigned int* bb = Bpk + ((size_t)(kt * CT2 + ctBase) * 32 + lane) * 8;
        #pragma unroll
        for (int ct = 0; ct < 4; ++ct) {
            AFrag b;
            b.q[0] = *(const uint4*)(bb + ct * 256);
            b.q[1] = *(const uint4*)(bb + ct * 256 + 4);
            acc[ct] = __builtin_amdgcn_wmma_f32_16x16x32_bf16(false, a.v, false, b.v,
                                                              (short)0, acc[ct], false, false);
        }
    }
    #pragma unroll
    for (int ct = 0; ct < 4; ++ct) {
        int col = (ctBase + ct) * 16 + ln;
        if (col < EMB) {
            float bv = bias[col];
            #pragma unroll
            for (int j = 0; j < 8; ++j) {
                int row = rowBase + 8 * g + j;
                if (row < N_NODES)
                    out[(size_t)row * EMB + col] = acc[ct][j] + bv;
            }
        }
    }
}

// ---------------- BN column reduce (sum, sumsq) ----------------
__global__ void k_bnreduce(const float* __restrict__ h, float* __restrict__ stats) {
    __shared__ float ss[256], sq[256];
    int col = blockIdx.x;
    float s = 0.f, q = 0.f;
    for (long long n = (long long)blockIdx.y * blockDim.x + threadIdx.x; n < N_NODES;
         n += (long long)gridDim.y * blockDim.x) {
        float v = h[n * EMB + col];
        s += v; q += v * v;
    }
    ss[threadIdx.x] = s; sq[threadIdx.x] = q;
    __syncthreads();
    for (int o = 128; o > 0; o >>= 1) {
        if (threadIdx.x < o) {
            ss[threadIdx.x] += ss[threadIdx.x + o];
            sq[threadIdx.x] += sq[threadIdx.x + o];
        }
        __syncthreads();
    }
    if (threadIdx.x == 0) {
        unsafeAtomicAdd(&stats[col], ss[0]);
        unsafeAtomicAdd(&stats[EMB + col], sq[0]);
    }
}

// ---------------- BN normalize (+ optional ReLU), in place ----------------
__global__ void k_bnapply(float* __restrict__ h, const float* __restrict__ stats,
                          const float* __restrict__ gamma, const float* __restrict__ beta,
                          int do_relu) {
    long long i = (long long)blockIdx.x * blockDim.x + threadIdx.x;
    if (i >= (long long)N_NODES * EMB) return;
    int d = (int)(i % EMB);
    const float invN = 1.f / (float)N_NODES;
    float mean = stats[d] * invN;
    float var  = stats[EMB + d] * invN - mean * mean;
    float v = (h[i] - mean) * rsqrtf(var + BN_EPS) * gamma[d] + beta[d];
    if (do_relu) v = v > 0.f ? v : 0.f;
    h[i] = v;
}

extern "C" void kernel_launch(void* const* d_in, const int* in_sizes, int n_in,
                              void* d_out, int out_size, void* d_ws, size_t ws_size,
                              hipStream_t stream) {
    const int*   x         = (const int*)d_in[0];
    const int*   edge_index= (const int*)d_in[1];
    const int*   edge_attr = (const int*)d_in[2];
    const float* atom_emb1 = (const float*)d_in[3];
    const float* atom_emb2 = (const float*)d_in[4];
    const float* ee1       = (const float*)d_in[5];
    const float* ee2       = (const float*)d_in[6];
    const float* W1        = (const float*)d_in[7];
    const float* b1        = (const float*)d_in[8];
    const float* W2        = (const float*)d_in[9];
    const float* b2        = (const float*)d_in[10];
    const float* gamma     = (const float*)d_in[11];
    const float* beta      = (const float*)d_in[12];
    float* h = (float*)d_out;

    char* ws = (char*)d_ws;
    size_t off = 0;
    float* agg = (float*)(ws + off);                    off += (size_t)NPAD * KP1 * sizeof(float);
    unsigned int* aggb = (unsigned int*)(ws + off);     off += (size_t)NPAD * KP1 * 2;
    unsigned short* hid = (unsigned short*)(ws + off);  off += (size_t)NPAD * COLP1 * 2;
    unsigned int* W1pk = (unsigned int*)(ws + off);     off += (size_t)KT1 * CT1 * 256 * 4;
    unsigned int* W2pk = (unsigned int*)(ws + off);     off += (size_t)KT2 * CT2 * 256 * 4;
    float* stats = (float*)(ws + off);                  off += 2 * EMB * sizeof(float);

    {
        long long tot = (long long)N_NODES * EMB;
        k_embed<<<(unsigned)((tot + 255) / 256), 256, 0, stream>>>(x, atom_emb1, atom_emb2, h);
    }
    for (int l = 0; l < NUM_LAYER; ++l) {
        hipMemsetAsync(agg, 0, (size_t)NPAD * KP1 * sizeof(float), stream);
        hipMemsetAsync(stats, 0, 2 * EMB * sizeof(float), stream);
        k_scatter<<<N_EDGES + N_NODES, 320, 0, stream>>>(
            edge_index, edge_attr, ee1 + (size_t)l * 6 * EMB, ee2 + (size_t)l * 3 * EMB, h, agg);
        {
            long long pairs = (long long)NPAD * KP1 / 2;
            k_convA<<<(unsigned)((pairs + 255) / 256), 256, 0, stream>>>(agg, aggb);
        }
        k_packW1<<<(KT1 * CT1 * 256 + 255) / 256, 256, 0, stream>>>(W1 + (size_t)l * EMB2 * EMB, W1pk);
        k_packW2<<<(KT2 * CT2 * 256 + 255) / 256, 256, 0, stream>>>(W2 + (size_t)l * EMB * EMB2, W2pk);
        dim3 g1(CT1 / 4, NPAD / 128);          // 10 x 782
        k_gemm1<<<g1, 256, 0, stream>>>((const unsigned short*)aggb, W1pk, b1 + l * EMB2, hid);
        dim3 g2(CT2 / 4, NPAD / 128);          // 5 x 782
        k_gemm2<<<g2, 256, 0, stream>>>(hid, W2pk, b2 + l * EMB, h);
        dim3 gr(EMB, 64);
        k_bnreduce<<<gr, 256, 0, stream>>>(h, stats);
        long long tot = (long long)N_NODES * EMB;
        k_bnapply<<<(unsigned)((tot + 255) / 256), 256, 0, stream>>>(
            h, stats, gamma + l * EMB, beta + l * EMB, (l < NUM_LAYER - 1) ? 1 : 0);
    }
}
